// MultiHeadAttention_936302870847
// MI455X (gfx1250) — matrix operand order repaired and checked by reading, not compiled, not run on hardware
//
#include <hip/hip_runtime.h>

// ---------------------------------------------------------------------------
// MI455X (gfx1250, wave32) fused multi-head causal attention.
//   proj GEMMs + flash attention + out GEMM, all on v_wmma_f32_16x16x32_f16.
// Compute-bound (~137 GFLOP vs <200MB HBM traffic at 23.3 TB/s): minimize
// non-WMMA instructions per WMMA. Softmax reductions use DPP16 butterflies
// (VALU) instead of ds_bpermute; all staging uses 128-bit loads; V staged
// transposed so P@V B-fragments read as ds_load_b128.
// ---------------------------------------------------------------------------

typedef __attribute__((ext_vector_type(16))) _Float16 v16h;
typedef __attribute__((ext_vector_type(8)))  float    v8f;

union FragH  { v16h v; unsigned int u[8]; _Float16 h[16]; };
union Half4  { unsigned long long u64; _Float16 h[4]; };
union Half8  { uint4 q; _Float16 h[8]; };

#define D_MODEL   1024
#define NUM_HEADS 16
#define D_K       64
#define BATCH     4
#define SEQ       2048
#define M_ROWS    (BATCH * SEQ)   // 8192

// ---- DPP16 butterfly reductions across the 16-lane row group -------------
// xor1 = quad_perm[1,0,3,2] (0xB1), xor2 = quad_perm[2,3,0,1] (0x4E),
// then ROW_HALF_MIRROR (0x141, ==xor4 once quads uniform) and
// ROW_MIRROR (0x140, ==xor8 once halves uniform). Pure VALU, no LDS.
template <int CTRL>
__device__ __forceinline__ float dpp_perm(float v) {
  return __int_as_float(__builtin_amdgcn_update_dpp(
      0, __float_as_int(v), CTRL, 0xF, 0xF, true));
}
__device__ __forceinline__ float row16_max(float v) {
  v = fmaxf(v, dpp_perm<0xB1>(v));
  v = fmaxf(v, dpp_perm<0x4E>(v));
  v = fmaxf(v, dpp_perm<0x141>(v));
  v = fmaxf(v, dpp_perm<0x140>(v));
  return v;
}
__device__ __forceinline__ float row16_sum(float v) {
  v += dpp_perm<0xB1>(v);
  v += dpp_perm<0x4E>(v);
  v += dpp_perm<0x141>(v);
  v += dpp_perm<0x140>(v);
  return v;
}

// ---- vectorized global->f16 loaders ---------------------------------------
__device__ __forceinline__ void load4(const float* p, Half4& out) {
  float4 f = *(const float4*)p;
  out.h[0] = (_Float16)f.x; out.h[1] = (_Float16)f.y;
  out.h[2] = (_Float16)f.z; out.h[3] = (_Float16)f.w;
}
__device__ __forceinline__ void load4(const _Float16* p, Half4& out) {
  uint2 u = *(const uint2*)p;
  *(uint2*)out.h = u;
}

// ---------------------------------------------------------------------------
// GEMM: out[M,1024] = A[M,1024] @ W[1024,1024]^T + bias
// Block: 256 threads = 8 waves; block tile 64(M) x 256(N); K step 32.
// Waves 2(M) x 4(N): each wave owns 32x64 = 2x4 WMMA accumulators
// (8 WMMA per 12 ds_load_b128 per K step; A fragments reused 4x).
// ---------------------------------------------------------------------------
template <typename AT, bool OUT_F32>
__global__ __launch_bounds__(256) void gemm_xwt_kernel(
    const AT* __restrict__ A, const float* __restrict__ W,
    const float* __restrict__ bias, void* __restrict__ out) {
  // stride 40 halves (80B): bank-spread, keeps b64/b128 accesses aligned
  __shared__ _Float16 aT[64 * 40];
  __shared__ _Float16 bT[256 * 40];

  const int tid  = threadIdx.x;
  const int lane = tid & 31;
  const int wid  = tid >> 5;
  const int wm   = wid >> 2;   // 0..1
  const int wn   = wid & 3;    // 0..3
  const int hl   = lane >> 4;
  const int ml   = lane & 15;

  const int blockM = blockIdx.x * 64;
  const int blockN = blockIdx.y * 256;

  v8f zero = {};
  v8f acc[2][4];
#pragma unroll
  for (int i = 0; i < 2; ++i)
#pragma unroll
    for (int j = 0; j < 4; ++j) acc[i][j] = zero;

  const int colLd4 = (tid & 7) * 4;  // 4-element column group within K step
  const int rowLd  = tid >> 3;       // 0..31

  for (int k0 = 0; k0 < 1024; k0 += 32) {
    // stage A 64x32 (2 float4/thread) and B 256x32 (8 float4/thread) as f16
#pragma unroll
    for (int rr = 0; rr < 2; ++rr) {
      int row = rowLd + rr * 32;
      Half4 p;
      load4(&A[(size_t)(blockM + row) * 1024 + k0 + colLd4], p);
      *(unsigned long long*)&aT[row * 40 + colLd4] = p.u64;
    }
#pragma unroll
    for (int rr = 0; rr < 8; ++rr) {
      int row = rowLd + rr * 32;
      Half4 p;
      load4(&W[(size_t)(blockN + row) * 1024 + k0 + colLd4], p);
      *(unsigned long long*)&bT[row * 40 + colLd4] = p.u64;
    }
    __syncthreads();

    // A fragment: 16x32 f16; VGPR j holds K pair (j<4?0:16)+(j&3)*2+hl*8
    FragH af[2], bf[4];
#pragma unroll
    for (int ms = 0; ms < 2; ++ms) {
      int row = wm * 32 + ms * 16 + ml;
#pragma unroll
      for (int j = 0; j < 8; ++j) {
        int k = ((j < 4) ? 0 : 16) + (j & 3) * 2 + hl * 8;
        af[ms].u[j] = *(const unsigned int*)&aT[row * 40 + k];
      }
    }
  // B fragment: 32x16 f16; col n = ml; VGPR j holds K pair (j<4?0:16)+(j&3)*2+hl*8
#pragma unroll
    for (int ns = 0; ns < 4; ++ns) {
      int col = wn * 64 + ns * 16 + ml;
#pragma unroll
      for (int j = 0; j < 8; ++j) {
        int k = ((j < 4) ? 0 : 16) + (j & 3) * 2 + hl * 8;
        bf[ns].u[j] = *(const unsigned int*)&bT[col * 40 + k];
      }
    }
#pragma unroll
    for (int ms = 0; ms < 2; ++ms)
#pragma unroll
      for (int ns = 0; ns < 4; ++ns)
        acc[ms][ns] = __builtin_amdgcn_wmma_f32_16x16x32_f16(
            false, af[ms].v, false, bf[ns].v, (short)0, acc[ms][ns], false,
            false);
    __syncthreads();
  }

  // epilogue: C layout row = r + 8*hl, col = ml
#pragma unroll
  for (int ms = 0; ms < 2; ++ms) {
#pragma unroll
    for (int ns = 0; ns < 4; ++ns) {
      int col = blockN + wn * 64 + ns * 16 + ml;
      float bv = bias[col];
#pragma unroll
      for (int r = 0; r < 8; ++r) {
        int row   = blockM + wm * 32 + ms * 16 + r + hl * 8;
        float val = acc[ms][ns][r] + bv;
        if constexpr (OUT_F32)
          ((float*)out)[(size_t)row * 1024 + col] = val;
        else
          ((_Float16*)out)[(size_t)row * 1024 + col] = (_Float16)val;
      }
    }
  }
}

// ---------------------------------------------------------------------------
// Flash attention, causal. One wave per 16-query tile; 4 waves per block with
// wave-private LDS (no block barriers -> causal trip-count divergence is OK).
// K staged row-major ([key][d], stride 72), V staged TRANSPOSED ([d][key],
// stride 40) so both score-B and V-B fragments read as contiguous b128.
// ---------------------------------------------------------------------------
__global__ __launch_bounds__(128) void attn_kernel(
    const _Float16* __restrict__ Q, const _Float16* __restrict__ K,
    const _Float16* __restrict__ V, _Float16* __restrict__ O) {
  // per wave: kT 32*72 + vT 64*40 + pT 16*32 = 5376 halves = 10752B; x4 waves
  __shared__ _Float16 smem[4][5376];

  const int tid  = threadIdx.x;
  const int lane = tid & 31;
  const int wid  = tid >> 5;
  const int hl   = lane >> 4;
  const int ml   = lane & 15;

  const int bh = blockIdx.x;  // 0..63
  const int b  = bh >> 4;
  const int h  = bh & 15;
  const int qt = blockIdx.y * 4 + wid;  // query tile 0..127
  const int m0 = qt * 16;

  _Float16* kT = smem[wid];                 // [key][d]  stride 72
  _Float16* vT = smem[wid] + 32 * 72;       // [d][key]  stride 40 (transposed)
  _Float16* pT = smem[wid] + 32 * 72 + 64 * 40;  // 16x32

  const _Float16* qBase = Q + (size_t)b * SEQ * D_MODEL + h * D_K;
  const _Float16* kBase = K + (size_t)b * SEQ * D_MODEL + h * D_K;
  const _Float16* vBase = V + (size_t)b * SEQ * D_MODEL + h * D_K;

  // Q fragments (16x32 each, d=0..31 / 32..63) straight from global
  FragH qa[2];
  {
    const _Float16* qrow = qBase + (size_t)(m0 + ml) * D_MODEL;
#pragma unroll
    for (int kk = 0; kk < 2; ++kk)
#pragma unroll
      for (int j = 0; j < 8; ++j) {
        int d = kk * 32 + ((j < 4) ? 0 : 16) + (j & 3) * 2 + hl * 8;
        qa[kk].u[j] = *(const unsigned int*)&qrow[d];
      }
  }

  v8f zero = {};
  float rmax[8], rsum[8];
  v8f oacc[4];
#pragma unroll
  for (int r = 0; r < 8; ++r) { rmax[r] = -1.0e30f; rsum[r] = 0.f; }
#pragma unroll
  for (int ds = 0; ds < 4; ++ds) oacc[ds] = zero;

  const int ktMax = (m0 + 15) >> 5;  // causal bound (uniform per wave)

  for (int kt = 0; kt <= ktMax; ++kt) {
    const int key0 = kt * 32;

    // prefetch next tile's rows (one cacheline per lane) -> global_prefetch_b8
    if (kt < ktMax) {
      __builtin_prefetch(&kBase[(size_t)(key0 + 32 + lane) * D_MODEL], 0, 0);
      __builtin_prefetch(&vBase[(size_t)(key0 + 32 + lane) * D_MODEL], 0, 0);
    }

    // stage tiles: 8 b128 global loads per lane each for K and V
#pragma unroll
    for (int i = 0; i < 8; ++i) {
      int oct = i * 32 + lane;      // 0..255 octets of 8 halves
      int key = oct >> 3;           // 0..31
      int d8  = (oct & 7) * 8;      // 0..56
      // K: row-major copy, b128 store
      *(uint4*)&kT[key * 72 + d8] =
          *(const uint4*)&kBase[(size_t)(key0 + key) * D_MODEL + d8];
      // V: transpose into [d][key] with b16 scatter stores
      Half8 vv;
      vv.q = *(const uint4*)&vBase[(size_t)(key0 + key) * D_MODEL + d8];
#pragma unroll
      for (int c = 0; c < 8; ++c) vT[(d8 + c) * 40 + key] = vv.h[c];
    }
    // wave-local visibility: DS ops from this wave complete in order
    asm volatile("s_wait_dscnt 0" ::: "memory");

    // scores: c[s] = Q(16x64) @ K_sub^T ; s=0 keys 0..15, s=1 keys 16..31
    v8f c[2];
#pragma unroll
    for (int s = 0; s < 2; ++s) {
      c[s] = zero;
      int keyl = s * 16 + ml;
#pragma unroll
      for (int kk = 0; kk < 2; ++kk) {
        FragH bfr;
#pragma unroll
        for (int j = 0; j < 8; ++j) {
          int d = kk * 32 + ((j < 4) ? 0 : 16) + (j & 3) * 2 + hl * 8;
          bfr.u[j] = *(const unsigned int*)&kT[keyl * 72 + d];
        }
        c[s] = __builtin_amdgcn_wmma_f32_16x16x32_f16(
            false, qa[kk].v, false, bfr.v, (short)0, c[s], false, false);
      }
    }

    // online softmax (rows r+8*hl, col ml per subtile); DPP reductions
    float corr[8];
#pragma unroll
    for (int r = 0; r < 8; ++r) {
      int mrow = m0 + r + hl * 8;
      float s0 = c[0][r] * 0.125f;  // 1/sqrt(64)
      float s1 = c[1][r] * 0.125f;
      if (key0 + ml > mrow)      s0 = -1.0e9f;  // causal (== tril reference)
      if (key0 + 16 + ml > mrow) s1 = -1.0e9f;
      float tmax = row16_max(fmaxf(s0, s1));
      float nm   = fmaxf(rmax[r], tmax);
      float cr   = __expf(rmax[r] - nm);
      rmax[r]    = nm;
      float p0   = __expf(s0 - nm);
      float p1   = __expf(s1 - nm);
      rsum[r]    = rsum[r] * cr + row16_sum(p0 + p1);
      corr[r]    = cr;
      int mr = r + hl * 8;
      pT[mr * 32 + ml]      = (_Float16)p0;
      pT[mr * 32 + 16 + ml] = (_Float16)p1;
    }
#pragma unroll
    for (int ds = 0; ds < 4; ++ds)
#pragma unroll
      for (int r = 0; r < 8; ++r) oacc[ds][r] *= corr[r];

    asm volatile("s_wait_dscnt 0" ::: "memory");

    // P as A fragment (16x32)
    FragH pa;
#pragma unroll
    for (int j = 0; j < 8; ++j) {
      int k = ((j < 4) ? 0 : 16) + (j & 3) * 2 + hl * 8;
      pa.u[j] = *(const unsigned int*)&pT[ml * 32 + k];
    }
    // O += P @ V : B fragment from transposed vT -> contiguous b128 reads
#pragma unroll
    for (int ds = 0; ds < 4; ++ds) {
      FragH vf;
#pragma unroll
      for (int j = 0; j < 8; ++j) {
        int k = ((j < 4) ? 0 : 16) + (j & 3) * 2 + hl * 8;  // key_local pair
        vf.u[j] = *(const unsigned int*)&vT[(ds * 16 + ml) * 40 + k];
      }
      oacc[ds] = __builtin_amdgcn_wmma_f32_16x16x32_f16(
          false, pa.v, false, vf.v, (short)0, oacc[ds], false, false);
    }
  }

  // normalize, store head-interleaved [B,S,H*Dk] for the Wo GEMM
  _Float16* oBase = O + (size_t)b * SEQ * D_MODEL + h * D_K;
#pragma unroll
  for (int r = 0; r < 8; ++r) {
    float inv = 1.0f / rsum[r];
    int m = m0 + r + hl * 8;
#pragma unroll
    for (int ds = 0; ds < 4; ++ds)
      oBase[(size_t)m * D_MODEL + ds * 16 + ml] =
          (_Float16)(oacc[ds][r] * inv);
  }
}

// ---------------------------------------------------------------------------
extern "C" void kernel_launch(void* const* d_in, const int* in_sizes, int n_in,
                              void* d_out, int out_size, void* d_ws,
                              size_t ws_size, hipStream_t stream) {
  const float* x  = (const float*)d_in[0];
  // d_in[1] (mask) is tril -> handled analytically in attn_kernel
  const float* Wq = (const float*)d_in[2];
  const float* bq = (const float*)d_in[3];
  const float* Wk = (const float*)d_in[4];
  const float* bk = (const float*)d_in[5];
  const float* Wv = (const float*)d_in[6];
  const float* bv = (const float*)d_in[7];
  const float* Wo = (const float*)d_in[8];
  const float* bo = (const float*)d_in[9];

  _Float16* qws  = (_Float16*)d_ws;  // 16 MB each
  _Float16* kws  = qws + (size_t)M_ROWS * D_MODEL;
  _Float16* vws  = kws + (size_t)M_ROWS * D_MODEL;
  _Float16* aows = vws + (size_t)M_ROWS * D_MODEL;

  dim3 ggrid(M_ROWS / 64, D_MODEL / 256);  // 128 x 4 blocks
  gemm_xwt_kernel<float, false><<<ggrid, 256, 0, stream>>>(x, Wq, bq, qws);
  gemm_xwt_kernel<float, false><<<ggrid, 256, 0, stream>>>(x, Wk, bk, kws);
  gemm_xwt_kernel<float, false><<<ggrid, 256, 0, stream>>>(x, Wv, bv, vws);

  attn_kernel<<<dim3(BATCH * NUM_HEADS, SEQ / 16 / 4), 128, 0, stream>>>(
      qws, kws, vws, aows);

  gemm_xwt_kernel<_Float16, true><<<ggrid, 256, 0, stream>>>(aows, Wo, bo,
                                                             d_out);
}
